// SimpleGCN_72206990181016
// MI455X (gfx1250) — matrix operand order repaired and checked by reading, not compiled, not run on hardware
//
#include <hip/hip_runtime.h>

typedef __attribute__((ext_vector_type(16))) _Float16 v16h;
typedef __attribute__((ext_vector_type(8)))  float    v8f;

#define N_NODES 100000
#define K_DIM   128

// ---------------- degree kernels ----------------
__global__ void gcn_deg_init(float* deg, int n) {
  int i = blockIdx.x * blockDim.x + threadIdx.x;
  if (i < n) deg[i] = 1.0f;  // self-loop contributes 1
}

__global__ void gcn_deg_accum(const int* __restrict__ dst, float* deg, int E) {
  int e = blockIdx.x * blockDim.x + threadIdx.x;
  if (e < E) atomicAdd(&deg[dst[e]], 1.0f);
}

__global__ void gcn_deg_fin(float* deg, int n) {
  int i = blockIdx.x * blockDim.x + threadIdx.x;
  if (i < n) deg[i] = rsqrtf(deg[i]);  // deg >= 1 always
}

// ---------------- pack W into fragment-major f16 image ----------------
// P[((ct*4+ks)*32 + lane)*16 + t] = W[(ks*32 + 16*(lane>>4) + t)][ct*16 + (lane&15)]
// This is contiguous k order (k = 16*(lane>>4) + t), the same order the A fragments use, so a
// lane's whole fragment is 16 contiguous halves (32B, one aligned LDS read).
template<int DOUT>
__global__ void gcn_pack_w(const float* __restrict__ W, _Float16* __restrict__ P) {
  int idx = blockIdx.x * blockDim.x + threadIdx.x;        // one thread per half
  if (idx >= K_DIM * DOUT) return;
  int t    = idx & 15;
  int lane = (idx >> 4) & 31;
  int f    = idx >> 9;
  int ks   = f & 3;
  int ct   = f >> 2;
  int row  = ks * 32 + 16 * (lane >> 4) + t;              // K index
  int col  = ct * 16 + (lane & 15);                       // N index
  P[idx] = (_Float16)W[row * DOUT + col];
}

// ---------------- WMMA GEMM: H[N x DOUT] = X[N x 128] * W[128 x DOUT] ----------------
// One wave32 computes a 16-row stripe across all DOUT/16 column tiles.
template<int DOUT>
__global__ __launch_bounds__(256) void gcn_gemm_wmma(
    const float* __restrict__ X, const _Float16* __restrict__ Pw,
    float* __restrict__ H, int nrows)
{
  constexpr int FR = (DOUT / 16) * 4;          // fragments (col tiles x K-steps)
  __shared__ uint4 Wl[FR * 32 * 2];            // 16 halves = 2 uint4 per (frag,lane)

  // Async bulk copy of the packed image into LDS (ASYNCcnt DMA path).
  // GV mode: per-lane 64-bit global addr (VGPR pair) -> per-lane LDS byte addr.
  {
    unsigned ldsbase = (unsigned)(unsigned long long)&Wl[0];  // LDS aperture: low 32 bits = LDS offset
    for (int i = threadIdx.x; i < FR * 32 * 2; i += 256) {
      unsigned ldsaddr = ldsbase + (unsigned)i * 16u;
      unsigned long long ga = (unsigned long long)(const char*)Pw + (unsigned long long)i * 16u;
      asm volatile("global_load_async_to_lds_b128 %0, %1, off"
                   :: "v"(ldsaddr), "v"(ga) : "memory");
    }
    asm volatile("s_wait_asynccnt 0" ::: "memory");  // drain ASYNCcnt before sharing
  }
  __syncthreads();

  int wid  = (int)((blockIdx.x * blockDim.x + threadIdx.x) >> 5);
  int lane = threadIdx.x & 31;
  int hh   = lane >> 4;     // 0: lanes 0-15, 1: lanes 16-31
  int mloc = lane & 15;
  int row0 = wid * 16;
  if (row0 >= nrows) return;  // wave-uniform: EXEC all-ones inside

  const float* xrow = X + (long)(row0 + mloc) * K_DIM;

  // A fragments for 4 K-steps of 32. Lane holds row M=mloc:
  //   a[0..15] = X[row][k0 + 16*hh + 0..15]
  v16h a[4];
#pragma unroll
  for (int ks = 0; ks < 4; ++ks) {
    int base = ks * 32 + 16 * hh;
    float4 p0 = *(const float4*)(xrow + base);
    float4 p1 = *(const float4*)(xrow + base + 4);
    float4 p2 = *(const float4*)(xrow + base + 8);
    float4 p3 = *(const float4*)(xrow + base + 12);
    a[ks][0]=(_Float16)p0.x; a[ks][1]=(_Float16)p0.y; a[ks][2]=(_Float16)p0.z; a[ks][3]=(_Float16)p0.w;
    a[ks][4]=(_Float16)p1.x; a[ks][5]=(_Float16)p1.y; a[ks][6]=(_Float16)p1.z; a[ks][7]=(_Float16)p1.w;
    a[ks][8]=(_Float16)p2.x; a[ks][9]=(_Float16)p2.y; a[ks][10]=(_Float16)p2.z; a[ks][11]=(_Float16)p2.w;
    a[ks][12]=(_Float16)p3.x; a[ks][13]=(_Float16)p3.y; a[ks][14]=(_Float16)p3.z; a[ks][15]=(_Float16)p3.w;
  }

#pragma unroll
  for (int ct = 0; ct < DOUT / 16; ++ct) {
    v8f c = {};
#pragma unroll
    for (int ks = 0; ks < 4; ++ks) {
      // whole B fragment: one aligned 32B LDS read (2x ds_load_b128)
      v16h b = *(const v16h*)&Wl[((ct * 4 + ks) * 32 + lane) * 2];
      c = __builtin_amdgcn_wmma_f32_16x16x32_f16(
          false, a[ks], false, b, (short)0, c, false, false);
    }
    // C/D layout: VGPR r -> row = r + 8*hh, col = mloc
    float* hp = H + (long)row0 * DOUT + ct * 16 + mloc;
#pragma unroll
    for (int r = 0; r < 8; ++r)
      hp[(long)(r + 8 * hh) * DOUT] = c[r];
  }
}

// ---------------- self-loop init: out[n,f] = h[n,f] * dinv[n]^2 ----------------
__global__ void gcn_agg_init(const float* __restrict__ h, const float* __restrict__ dinv,
                             float* __restrict__ out, long tot, int shift) {
  long i = (long)blockIdx.x * blockDim.x + threadIdx.x;
  if (i < tot) {
    int node = (int)(i >> shift);
    float s = dinv[node];
    out[i] = h[i] * s * s;
  }
}

// ---------------- edge scatter: wave32 per 4 edges, prefetch + f32 L2 atomics ----------------
template<int D>
__global__ __launch_bounds__(256) void gcn_scatter(
    const float* __restrict__ h, const int* __restrict__ src, const int* __restrict__ dst,
    const float* __restrict__ dinv, float* __restrict__ out, int E)
{
  constexpr int EPW = 4;                 // edges per wave
  constexpr int VPT = D / 32;            // floats per lane (4 or 2)
  int wid  = (int)((blockIdx.x * blockDim.x + threadIdx.x) >> 5);
  int lane = threadIdx.x & 31;
  int e0 = wid * EPW;
  if (e0 >= E) return;                   // wave-uniform
  int ne = E - e0; if (ne > EPW) ne = EPW;

  int ss[EPW], dd[EPW];
  for (int i = 0; i < ne; ++i) { ss[i] = src[e0 + i]; dd[i] = dst[e0 + i]; }
  // prefetch upcoming source rows (lowers to global_prefetch_b8)
  for (int i = 0; i < ne; ++i)
    __builtin_prefetch(h + (long)ss[i] * D + lane * VPT, 0, 0);

  for (int i = 0; i < ne; ++i) {
    float nrm = dinv[ss[i]] * dinv[dd[i]];
    const float* hs = h + (long)ss[i] * D + lane * VPT;
    float* od = out + (long)dd[i] * D + lane * VPT;
    if (VPT == 4) {
      float4 v = *(const float4*)hs;
      atomicAdd(&od[0], v.x * nrm);
      atomicAdd(&od[1], v.y * nrm);
      atomicAdd(&od[2], v.z * nrm);
      atomicAdd(&od[3], v.w * nrm);
    } else {
      float2 v = *(const float2*)hs;
      atomicAdd(&od[0], v.x * nrm);
      atomicAdd(&od[1], v.y * nrm);
    }
  }
}

// ---------------- bias + relu (layer 1 epilogue, D=128) ----------------
__global__ void gcn_bias_relu(float* h, const float* __restrict__ b, long tot) {
  long i = (long)blockIdx.x * blockDim.x + threadIdx.x;
  if (i < tot) {
    float v = h[i] + b[i & 127];
    h[i] = v > 0.0f ? v : 0.0f;
  }
}

// ---------------- bias + log_softmax over 64 cols: wave32 per node ----------------
__global__ __launch_bounds__(256) void gcn_bias_lsm(float* y, const float* __restrict__ b, int n) {
  int wid  = (int)((blockIdx.x * blockDim.x + threadIdx.x) >> 5);
  int lane = threadIdx.x & 31;
  if (wid >= n) return;
  float* row = y + (long)wid * 64;
  float v0 = row[lane]      + b[lane];
  float v1 = row[lane + 32] + b[lane + 32];
  float m = fmaxf(v0, v1);
#pragma unroll
  for (int off = 16; off; off >>= 1) m = fmaxf(m, __shfl_xor(m, off, 32));
  float s = expf(v0 - m) + expf(v1 - m);
#pragma unroll
  for (int off = 16; off; off >>= 1) s += __shfl_xor(s, off, 32);
  float ls = logf(s);
  row[lane]      = v0 - m - ls;
  row[lane + 32] = v1 - m - ls;
}

// ---------------- driver ----------------
extern "C" void kernel_launch(void* const* d_in, const int* in_sizes, int n_in,
                              void* d_out, int out_size, void* d_ws, size_t ws_size,
                              hipStream_t stream) {
  const float* x  = (const float*)d_in[0];
  const int*   ei = (const int*)d_in[1];   // [2, E] flat: src then dst
  const float* W1 = (const float*)d_in[2];
  const float* b1 = (const float*)d_in[3];
  const float* W2 = (const float*)d_in[4];
  const float* b2 = (const float*)d_in[5];
  float* out = (float*)d_out;

  int E = in_sizes[1] / 2;
  const int* src = ei;
  const int* dst = ei + E;

  // workspace layout (floats); needs ~103.3 MB
  float* ws   = (float*)d_ws;
  float* dinv = ws;                                 // 100000 (padded to 100352)
  float* buf1 = ws + 100352;                        // N*128 (h1, then h2 reuses)
  float* buf2 = buf1 + (long)N_NODES * 128;         // N*128 (agg1 / relu'd h1)
  _Float16* pw1 = (_Float16*)(buf2 + (long)N_NODES * 128);  // 16384 halves
  _Float16* pw2 = pw1 + K_DIM * 128;                        // 8192 halves

  const int TPB = 256;

  // degrees -> dinv
  gcn_deg_init <<<(N_NODES + TPB - 1) / TPB, TPB, 0, stream>>>(dinv, N_NODES);
  gcn_deg_accum<<<(E + TPB - 1) / TPB,       TPB, 0, stream>>>(dst, dinv, E);
  gcn_deg_fin  <<<(N_NODES + TPB - 1) / TPB, TPB, 0, stream>>>(dinv, N_NODES);

  // pack weights into WMMA B-operand fragment images
  gcn_pack_w<128><<<(K_DIM * 128 + TPB - 1) / TPB, TPB, 0, stream>>>(W1, pw1);
  gcn_pack_w<64> <<<(K_DIM * 64  + TPB - 1) / TPB, TPB, 0, stream>>>(W2, pw2);

  int tiles = N_NODES / 16;            // 6250, exact
  int gblks = (tiles + 7) / 8;         // 8 waves per block
  int eblks = (E + 31) / 32;           // 8 waves/block, 4 edges/wave

  // ---- layer 1 ----
  gcn_gemm_wmma<128><<<gblks, TPB, 0, stream>>>(x, pw1, buf1, N_NODES);
  long tot1 = (long)N_NODES * 128;
  gcn_agg_init<<<(int)((tot1 + TPB - 1) / TPB), TPB, 0, stream>>>(buf1, dinv, buf2, tot1, 7);
  gcn_scatter<128><<<eblks, TPB, 0, stream>>>(buf1, src, dst, dinv, buf2, E);
  gcn_bias_relu<<<(int)((tot1 + TPB - 1) / TPB), TPB, 0, stream>>>(buf2, b1, tot1);

  // ---- layer 2 ----
  gcn_gemm_wmma<64><<<gblks, TPB, 0, stream>>>(buf2, pw2, buf1, N_NODES);
  long tot2 = (long)N_NODES * 64;
  gcn_agg_init<<<(int)((tot2 + TPB - 1) / TPB), TPB, 0, stream>>>(buf1, dinv, out, tot2, 6);
  gcn_scatter<64><<<eblks, TPB, 0, stream>>>(buf1, src, dst, dinv, out, E);
  gcn_bias_lsm<<<(N_NODES + 7) / 8, TPB, 0, stream>>>(out, b2, N_NODES);
}
